// TopKRouter_60198261621196
// MI455X (gfx1250) — matrix operand order repaired and checked by reading, not compiled, not run on hardware
//
#include <hip/hip_runtime.h>
#include <hip/hip_bf16.h>

// Problem constants (from the reference).
#define N_TOK  131072
#define DIM    2048
#define NEXP   64
#define KSLAB  256                  // K columns of gate_w staged in LDS per slab
#define WG_TOK 256                  // tokens per workgroup (8 waves x 32 tokens)
#define NWG    (N_TOK / WG_TOK)     // 512

typedef __attribute__((ext_vector_type(16))) __bf16 v16bf;
typedef __attribute__((ext_vector_type(8)))  float  v8f;

union BF16Frag { v16bf v; uint4 q[2]; };

__global__ __launch_bounds__(256) void router_main(
    const float* __restrict__ x,          // [N_TOK, DIM] fp32
    const float* __restrict__ gw,         // [NEXP, DIM]  fp32
    const float* __restrict__ gb,         // [NEXP]       fp32
    float* __restrict__ out,              // [N_TOK, NEXP] routing weights
    float* __restrict__ usage_part)       // [NWG, NEXP]  per-WG expert usage
{
    // 64KB region: first used as bf16 gate_w slab [64][KSLAB] (32KB),
    // later reused as fp32 logits/weights [WG_TOK][NEXP] (64KB).
    __shared__ __align__(16) unsigned char raw[WG_TOK * NEXP * 4];
    __shared__ float sbias[NEXP];
    __shared__ float psum[4][NEXP];

    __bf16* lds_w = (__bf16*)raw;   // [NEXP][KSLAB] bf16
    float*  lds_l = (float*)raw;    // [WG_TOK][NEXP] fp32

    const int t     = threadIdx.x;
    const int lane  = t & 31;
    const int wave  = t >> 5;
    const int lmod  = lane & 15;
    const int lhalf = lane >> 4;          // 0 or 1
    const int wg    = blockIdx.x;
    const int tok0  = wg * WG_TOK;        // first global token of this WG
    const int wtok  = tok0 + wave * 32;   // first global token of this wave

    if (t < NEXP) sbias[t] = gb[t];

    v8f acc[2][4];
    #pragma unroll
    for (int tt = 0; tt < 2; ++tt)
        #pragma unroll
        for (int et = 0; et < 4; ++et)
            #pragma unroll
            for (int r = 0; r < 8; ++r) acc[tt][et][r] = 0.0f;

    for (int ks = 0; ks < DIM / KSLAB; ++ks) {
        // ---- Stage gate_w slab (all 64 experts, KSLAB columns) as bf16 in LDS.
        // 64*256 fp32 elements; 256 threads x 16 float4 each. Coalesced global reads.
        #pragma unroll
        for (int i = 0; i < 16; ++i) {
            const int eb = i * 4 + (t >> 6);     // expert row 0..63
            const int c  = (t & 63) * 4;         // column within slab, mult of 4
            const float4 f = *(const float4*)(gw + (size_t)eb * DIM + ks * KSLAB + c);
            union { ushort4 s; __bf16 h[4]; } pk;
            pk.h[0] = (__bf16)f.x; pk.h[1] = (__bf16)f.y;
            pk.h[2] = (__bf16)f.z; pk.h[3] = (__bf16)f.w;
            *(ushort4*)(lds_w + eb * KSLAB + c) = pk.s;   // 8B aligned DS store
        }
        __syncthreads();

        // ---- Inner K loop: 8 steps of K=32.
        for (int kk = 0; kk < KSLAB; kk += 32) {
            // A fragments (fp32 from global, cvt->bf16).
            // 16-bit A 16x32 layout: lanes 0-15 hold K {0..7,16..23}, lanes 16-31 K {8..15,24..31}.
            BF16Frag a[2];
            const int ka = ks * KSLAB + kk + (lhalf << 3);
            #pragma unroll
            for (int tt = 0; tt < 2; ++tt) {
                const float* xr = x + (size_t)(wtok + tt * 16 + lmod) * DIM + ka;
                const float4 f0 = *(const float4*)(xr + 0);
                const float4 f1 = *(const float4*)(xr + 4);
                const float4 f2 = *(const float4*)(xr + 16);
                const float4 f3 = *(const float4*)(xr + 20);
                a[tt].v[0]  = (__bf16)f0.x; a[tt].v[1]  = (__bf16)f0.y;
                a[tt].v[2]  = (__bf16)f0.z; a[tt].v[3]  = (__bf16)f0.w;
                a[tt].v[4]  = (__bf16)f1.x; a[tt].v[5]  = (__bf16)f1.y;
                a[tt].v[6]  = (__bf16)f1.z; a[tt].v[7]  = (__bf16)f1.w;
                a[tt].v[8]  = (__bf16)f2.x; a[tt].v[9]  = (__bf16)f2.y;
                a[tt].v[10] = (__bf16)f2.z; a[tt].v[11] = (__bf16)f2.w;
                a[tt].v[12] = (__bf16)f3.x; a[tt].v[13] = (__bf16)f3.y;
                a[tt].v[14] = (__bf16)f3.z; a[tt].v[15] = (__bf16)f3.w;
            }
            // B fragments from LDS (bf16, already converted).
            // 16-bit B 32x16 layout: lane = column (lane&15); lanes 0-15 K {0..7,16..23}, lanes 16-31 K {8..15,24..31}.
            BF16Frag b[4];
            const int kb = kk + (lhalf << 3);
            #pragma unroll
            for (int et = 0; et < 4; ++et) {
                const uint4* p = (const uint4*)(lds_w + (et * 16 + lmod) * KSLAB + kb);
                b[et].q[0] = p[0];
                b[et].q[1] = p[2];
            }
            #pragma unroll
            for (int tt = 0; tt < 2; ++tt)
                #pragma unroll
                for (int et = 0; et < 4; ++et)
                    acc[tt][et] = __builtin_amdgcn_wmma_f32_16x16x32_bf16(
                        false, a[tt].v, false, b[et].v,
                        (short)0, acc[tt][et], false, false);
        }
        __syncthreads();   // all reads of lds_w done before next slab overwrite
    }

    // ---- Spill logits to LDS [WG_TOK][NEXP].
    // C/D 16x16 f32 layout: VGPR r holds M=r (lanes 0-15) and M=r+8 (lanes 16-31); N = lane&15.
    #pragma unroll
    for (int tt = 0; tt < 2; ++tt)
        #pragma unroll
        for (int et = 0; et < 4; ++et)
            #pragma unroll
            for (int r = 0; r < 8; ++r) {
                const int tl = wave * 32 + tt * 16 + r + 8 * lhalf;
                const int e  = et * 16 + lmod;
                lds_l[tl * NEXP + e] = acc[tt][et][r];
            }
    __syncthreads();

    // ---- One thread per token: add bias, top-2 (first-index-wins ties), softmax(2),
    //      overwrite own LDS row with dense routing weights.
    {
        float m1 = -1e30f, m2 = -1e30f;
        int   i1 = 0, i2 = 0;
        const float4* row4 = (const float4*)(lds_l + t * NEXP);
        #pragma unroll
        for (int g = 0; g < 16; ++g) {
            const float4 v = row4[g];
            const float vv[4] = { v.x, v.y, v.z, v.w };
            #pragma unroll
            for (int c = 0; c < 4; ++c) {
                const int   e   = g * 4 + c;
                const float val = vv[c] + sbias[e];
                if (val > m1)      { m2 = m1; i2 = i1; m1 = val; i1 = e; }
                else if (val > m2) { m2 = val; i2 = e; }
            }
        }
        const float e2 = __expf(m2 - m1);
        const float w1 = 1.0f / (1.0f + e2);
        const float w2 = e2 * w1;
        float4* wr = (float4*)(lds_l + t * NEXP);
        #pragma unroll
        for (int g = 0; g < 16; ++g) {
            float4 z; float zz[4];
            #pragma unroll
            for (int c = 0; c < 4; ++c) {
                const int e = g * 4 + c;
                zz[c] = (e == i1) ? w1 : ((e == i2) ? w2 : 0.0f);
            }
            z.x = zz[0]; z.y = zz[1]; z.z = zz[2]; z.w = zz[3];
            wr[g] = z;
        }
    }
    __syncthreads();

    // ---- Coalesced writeback of 256x64 routing weights.
    {
        const float4* lf4 = (const float4*)lds_l;
        float4* of4 = (float4*)(out + (size_t)tok0 * NEXP);
        #pragma unroll
        for (int i = 0; i < 16; ++i)
            of4[i * 256 + t] = lf4[i * 256 + t];
    }

    // ---- Deterministic per-WG expert-usage column sums (fixed summation order).
    {
        const int e  = t & 63;
        const int r0 = (t >> 6) * 64;     // 4 threads per expert, 64 rows each
        float s = 0.0f;
        for (int j = 0; j < 64; ++j)
            s += lds_l[(r0 + j) * NEXP + e];
        psum[t >> 6][e] = s;
    }
    __syncthreads();
    if (t < NEXP) {
        const float u = ((psum[0][t] + psum[1][t]) + psum[2][t]) + psum[3][t];
        usage_part[wg * NEXP + t] = u;
    }
}

// Tiny reduction kernel: deterministic loss from per-WG partials.
__global__ __launch_bounds__(64) void router_loss(
    const float* __restrict__ usage_part, float* __restrict__ out_loss)
{
    __shared__ float u[NEXP];
    const int t = threadIdx.x;
    float s = 0.0f;
    for (int w = 0; w < NWG; ++w)
        s += usage_part[w * NEXP + t];
    u[t] = s;
    __syncthreads();
    if (t == 0) {
        float total = 0.0f;
        for (int e = 0; e < NEXP; ++e) total += u[e];
        const float inv = 1.0f / total;
        float loss = 0.0f;
        for (int e = 0; e < NEXP; ++e) {
            const float d = u[e] * inv - (1.0f / (float)NEXP);
            loss += d * d;
        }
        out_loss[0] = loss;
    }
}

extern "C" void kernel_launch(void* const* d_in, const int* in_sizes, int n_in,
                              void* d_out, int out_size, void* d_ws, size_t ws_size,
                              hipStream_t stream) {
    const float* x  = (const float*)d_in[0];
    const float* gw = (const float*)d_in[1];
    const float* gb = (const float*)d_in[2];
    // d_in[3] is k == 2; top-2 is hardcoded (K from the reference shape).
    float* out        = (float*)d_out;
    float* usage_part = (float*)d_ws;    // NWG * NEXP floats, fully rewritten each call

    router_main<<<NWG, 256, 0, stream>>>(x, gw, gb, out, usage_part);
    router_loss<<<1, 64, 0, stream>>>(usage_part, out + (size_t)N_TOK * NEXP);
}
